// AttentionLayer_48335561949514
// MI455X (gfx1250) — matrix operand order repaired and checked by reading, not compiled, not run on hardware
//
#include <hip/hip_runtime.h>
#include <hip/hip_bf16.h>

// ---------------------------------------------------------------------------
// Attention layer for MI455X (gfx1250), f16 WMMA path with f32 accumulation.
// B=2, L=2048, D=1024, H=16, Hd=64.
// Round 2: software-pipelined (double-buffered) fragment loads so WMMAs are
// not gated by s_wait_loadcnt 0; 32x64 per-wave GEMM tiles for higher AI.
// ---------------------------------------------------------------------------

typedef __attribute__((ext_vector_type(16))) _Float16 v16h;
typedef __attribute__((ext_vector_type(8)))  _Float16 h8;
typedef __attribute__((ext_vector_type(4)))  _Float16 h4;
typedef __attribute__((ext_vector_type(8)))  float    v8f;

#define D_MODEL   1024
#define L_SEQ     2048
#define N_HEADS   16
#define HEAD_DIM  64
#define SCALE_F   0.125f   /* 64^-0.5 */

__device__ __forceinline__ v8f wmma16(v16h a, v16h b, v8f c) {
    // D = A(16x32 f16) * B(32x16 f16) + C(16x16 f32)
    return __builtin_amdgcn_wmma_f32_16x16x32_f16(
        /*neg_a=*/false, a, /*neg_b=*/false, b,
        /*c_mod=*/(short)0, c, /*reuse_a=*/false, /*reuse_b=*/false);
}

__device__ __forceinline__ v16h joinh(h8 lo, h8 hi) {
    v16h r;
#pragma unroll
    for (int i = 0; i < 8; ++i) { r[i] = lo[i]; r[i + 8] = hi[i]; }
    return r;
}

// A fragment: row-major f16 source, row stride `ld`, two 8-half runs at +0/+16
__device__ __forceinline__ v16h load_afrag(const _Float16* p) {
    return joinh(*(const h8*)p, *(const h8*)(p + 16));
}
// B fragment: two 8-half runs at +0/+16
__device__ __forceinline__ v16h load_bfrag(const _Float16* p) {
    return joinh(*(const h8*)p, *(const h8*)(p + 16));
}

// --------------------------------------------------------------------------
// f32 -> f16 conversion (4 elements / thread; all sizes are multiples of 4)
// --------------------------------------------------------------------------
__global__ __launch_bounds__(256)
void cvt_f32_f16_kernel(const float* __restrict__ in, _Float16* __restrict__ out, int n) {
    int i = (blockIdx.x * 256 + threadIdx.x) * 4;
    if (i < n) {
        float4 v = *(const float4*)(in + i);
        h4 o;
        o[0] = (_Float16)v.x; o[1] = (_Float16)v.y;
        o[2] = (_Float16)v.z; o[3] = (_Float16)v.w;
        *(h4*)(out + i) = o;
    }
}

// --------------------------------------------------------------------------
// QKV GEMM:  C[4096,3072] = Xh[4096,1024] @ Wqkv^T  + bias
// Wave tile: 32(M) x 64(N); 8 waves stacked on M per block.
// Double-buffered fragments; K-wrap keeps the pipeline branch-free.
// --------------------------------------------------------------------------
__global__ __launch_bounds__(256)
void qkv_gemm_kernel(const _Float16* __restrict__ Xh,
                     const _Float16* __restrict__ Wh,   // [3072,1024]
                     const float*    __restrict__ bias, // [3072]
                     _Float16* __restrict__ Qh,
                     _Float16* __restrict__ Kh,
                     _Float16* __restrict__ Vt) {
    const int lane  = threadIdx.x & 31;
    const int wave  = threadIdx.x >> 5;
    const int col   = lane & 15;
    const int hi    = lane >> 4;
    const int Mbase = blockIdx.y * 256 + wave * 32;
    const int Nbase = blockIdx.x * 64;

    const _Float16* arow0 = Xh + (size_t)(Mbase + col) * D_MODEL + hi * 8;
    const _Float16* arow1 = arow0 + (size_t)16 * D_MODEL;
    const _Float16* brow[4];
#pragma unroll
    for (int nt = 0; nt < 4; ++nt)
        brow[nt] = Wh + (size_t)(Nbase + nt * 16 + col) * D_MODEL + hi * 8;

    v8f acc[2][4] = {{v8f{}, v8f{}, v8f{}, v8f{}}, {v8f{}, v8f{}, v8f{}, v8f{}}};

    v16h a_cur[2], b_cur[4], a_nxt[2], b_nxt[4];
    a_cur[0] = load_afrag(arow0);
    a_cur[1] = load_afrag(arow1);
#pragma unroll
    for (int nt = 0; nt < 4; ++nt) b_cur[nt] = load_bfrag(brow[nt]);

#pragma unroll 2
    for (int k = 0; k < D_MODEL; k += 32) {
        const int kn = (k + 32) & (D_MODEL - 1);   // wrap: branch-free pipeline
        // issue next-iteration loads first (fly in WMMA shadow)
        a_nxt[0] = load_afrag(arow0 + kn);
        a_nxt[1] = load_afrag(arow1 + kn);
#pragma unroll
        for (int nt = 0; nt < 4; ++nt) b_nxt[nt] = load_bfrag(brow[nt] + kn);
        __builtin_prefetch(arow0 + ((k + 128) & (D_MODEL - 1)), 0, 0);

#pragma unroll
        for (int nt = 0; nt < 4; ++nt) acc[0][nt] = wmma16(a_cur[0], b_cur[nt], acc[0][nt]);
#pragma unroll
        for (int nt = 0; nt < 4; ++nt) acc[1][nt] = wmma16(a_cur[1], b_cur[nt], acc[1][nt]);

        a_cur[0] = a_nxt[0]; a_cur[1] = a_nxt[1];
#pragma unroll
        for (int nt = 0; nt < 4; ++nt) b_cur[nt] = b_nxt[nt];
    }

    // Epilogue: C layout -> lane holds (M = mt*16 + r + 8*hi, N = nt*16 + col)
#pragma unroll
    for (int nt = 0; nt < 4; ++nt) {
        const int n     = Nbase + nt * 16 + col;
        const float bv  = bias[n];
        const int which = n >> 10;           // 0:Q 1:K 2:V
        const int rem   = n & 1023;
        const int h     = rem >> 6;
        const int hd    = rem & 63;
#pragma unroll
        for (int mt = 0; mt < 2; ++mt) {
#pragma unroll
            for (int r = 0; r < 8; ++r) {
                const int token = Mbase + mt * 16 + r + hi * 8;
                const int b     = token >> 11;       // /L_SEQ
                const int t     = token & (L_SEQ - 1);
                const _Float16 hv = (_Float16)(acc[mt][nt][r] + bv);
                const size_t bh = (size_t)(b * N_HEADS + h);
                if (which == 0)      Qh[(bh * L_SEQ + t) * HEAD_DIM + hd] = hv;
                else if (which == 1) Kh[(bh * L_SEQ + t) * HEAD_DIM + hd] = hv;
                else                 Vt[(bh * HEAD_DIM + hd) * L_SEQ + t] = hv;
            }
        }
    }
}

// --------------------------------------------------------------------------
// Flash attention: one wave = one (b,h, 16-query) tile; 32-key steps.
// K fragments double-buffered across key tiles; V fragments issued at loop
// top so they load during softmax VALU work. LDS bounce for P.
// --------------------------------------------------------------------------
#define P_STRIDE 40   // halves per row: 80B -> conflict-free b128 reads, 16B aligned

__global__ __launch_bounds__(128)
void attn_kernel(const _Float16* __restrict__ Qh,
                 const _Float16* __restrict__ Kh,
                 const _Float16* __restrict__ Vt,
                 _Float16* __restrict__ Oh) {
    __shared__ _Float16 Plds[4][16 * P_STRIDE];

    const int lane  = threadIdx.x & 31;
    const int wave  = threadIdx.x >> 5;
    const int col   = lane & 15;
    const int hi    = lane >> 4;
    const int bh    = blockIdx.y;                    // b*H + h
    const int qbase = blockIdx.x * 64 + wave * 16;

    const _Float16* Qb = Qh + (size_t)bh * L_SEQ * HEAD_DIM;
    const _Float16* Kb = Kh + (size_t)bh * L_SEQ * HEAD_DIM;
    const _Float16* Vb = Vt + (size_t)bh * HEAD_DIM * L_SEQ;
    _Float16* Pw = &Plds[wave][0];

    // Preload Q tile A-fragments (Hd = 64 -> two K=32 fragments)
    v16h aq[2];
#pragma unroll
    for (int kk = 0; kk < 2; ++kk)
        aq[kk] = load_afrag(Qb + (size_t)(qbase + col) * HEAD_DIM + kk * 32 + hi * 8);

    // K fragment pointers: kf[ks][kk] = keys (kt + ks*16 .. +15), hd chunk kk
    const _Float16* kbase0 = Kb + (size_t)col * HEAD_DIM + hi * 8;
    v16h kf[2][2], kf_n[2][2];
#pragma unroll
    for (int ks = 0; ks < 2; ++ks)
#pragma unroll
        for (int kk = 0; kk < 2; ++kk)
            kf[ks][kk] = load_bfrag(kbase0 + (size_t)ks * 16 * HEAD_DIM + kk * 32);

    float m[8], lsum[8];
    v8f o[4] = {v8f{}, v8f{}, v8f{}, v8f{}};
#pragma unroll
    for (int r = 0; r < 8; ++r) { m[r] = -INFINITY; lsum[r] = 0.0f; }

    for (int kt = 0; kt < L_SEQ; kt += 32) {
        // ---- V fragments for THIS tile: issue early, consumed after softmax
        v16h vf[4];
#pragma unroll
        for (int nt = 0; nt < 4; ++nt)
            vf[nt] = load_bfrag(Vb + (size_t)(nt * 16 + col) * L_SEQ + kt + hi * 8);

        // ---- scores S = Q @ K^T with current K fragments
        v8f s0 = {}, s1 = {};
#pragma unroll
        for (int kk = 0; kk < 2; ++kk) {
            s0 = wmma16(aq[kk], kf[0][kk], s0);
            s1 = wmma16(aq[kk], kf[1][kk], s1);
        }

        // ---- K fragments for NEXT tile (wrap on last iteration; discarded)
        const int ktn = (kt + 32) & (L_SEQ - 1);
#pragma unroll
        for (int ks = 0; ks < 2; ++ks)
#pragma unroll
            for (int kk = 0; kk < 2; ++kk)
                kf_n[ks][kk] = load_bfrag(kbase0 + (size_t)(ktn + ks * 16) * HEAD_DIM + kk * 32);

        // ---- online softmax
        float rmax[8];
#pragma unroll
        for (int r = 0; r < 8; ++r) {
            s0[r] *= SCALE_F; s1[r] *= SCALE_F;
            rmax[r] = fmaxf(s0[r], s1[r]);
        }
#pragma unroll
        for (int off = 1; off < 16; off <<= 1) {
#pragma unroll
            for (int r = 0; r < 8; ++r)
                rmax[r] = fmaxf(rmax[r], __shfl_xor(rmax[r], off, 32));
        }
        float p0[8], p1[8], rs[8], alpha[8];
#pragma unroll
        for (int r = 0; r < 8; ++r) {
            const float nm = fmaxf(m[r], rmax[r]);
            alpha[r] = __expf(m[r] - nm);
            p0[r]    = __expf(s0[r] - nm);
            p1[r]    = __expf(s1[r] - nm);
            rs[r]    = p0[r] + p1[r];
            m[r]     = nm;
        }
#pragma unroll
        for (int off = 1; off < 16; off <<= 1) {
#pragma unroll
            for (int r = 0; r < 8; ++r)
                rs[r] += __shfl_xor(rs[r], off, 32);
        }
#pragma unroll
        for (int r = 0; r < 8; ++r) lsum[r] = lsum[r] * alpha[r] + rs[r];
#pragma unroll
        for (int nt = 0; nt < 4; ++nt)
#pragma unroll
            for (int r = 0; r < 8; ++r) o[nt][r] *= alpha[r];

        // ---- store P (C layout) into LDS as f16
#pragma unroll
        for (int r = 0; r < 8; ++r) {
            const int rr = r + hi * 8;
            Pw[rr * P_STRIDE + col]      = (_Float16)p0[r];
            Pw[rr * P_STRIDE + 16 + col] = (_Float16)p1[r];
        }
        __syncthreads();

        // ---- reload P in A layout (two contiguous 8-half runs per lane)
        const _Float16* pr = Pw + col * P_STRIDE + hi * 8;
        const v16h ap = joinh(*(const h8*)pr, *(const h8*)(pr + 16));

        // ---- O += P @ V  (V fragments loaded at loop top)
#pragma unroll
        for (int nt = 0; nt < 4; ++nt)
            o[nt] = wmma16(ap, vf[nt], o[nt]);
        __syncthreads();

        // rotate K double buffer
#pragma unroll
        for (int ks = 0; ks < 2; ++ks)
#pragma unroll
            for (int kk = 0; kk < 2; ++kk)
                kf[ks][kk] = kf_n[ks][kk];
    }

    // ---- normalize + store f16 row-major [token, D]
    const int b = bh >> 4;
    const int h = bh & 15;
    float inv[8];
#pragma unroll
    for (int r = 0; r < 8; ++r) inv[r] = 1.0f / lsum[r];
#pragma unroll
    for (int nt = 0; nt < 4; ++nt) {
#pragma unroll
        for (int r = 0; r < 8; ++r) {
            const int t = qbase + r + hi * 8;
            Oh[((size_t)(b * L_SEQ + t)) * D_MODEL + h * HEAD_DIM + nt * 16 + col] =
                (_Float16)(o[nt][r] * inv[r]);
        }
    }
}

// --------------------------------------------------------------------------
// Output projection: out[4096,1024] = Oh @ proj_w^T + proj_b   (f32 out)
// Same pipelined 32x64 wave tile as the QKV GEMM.
// --------------------------------------------------------------------------
__global__ __launch_bounds__(256)
void proj_gemm_kernel(const _Float16* __restrict__ Oh,
                      const _Float16* __restrict__ Wh,   // [1024,1024]
                      const float*    __restrict__ bias, // [1024]
                      float* __restrict__ out) {
    const int lane  = threadIdx.x & 31;
    const int wave  = threadIdx.x >> 5;
    const int col   = lane & 15;
    const int hi    = lane >> 4;
    const int Mbase = blockIdx.y * 256 + wave * 32;
    const int Nbase = blockIdx.x * 64;

    const _Float16* arow0 = Oh + (size_t)(Mbase + col) * D_MODEL + hi * 8;
    const _Float16* arow1 = arow0 + (size_t)16 * D_MODEL;
    const _Float16* brow[4];
#pragma unroll
    for (int nt = 0; nt < 4; ++nt)
        brow[nt] = Wh + (size_t)(Nbase + nt * 16 + col) * D_MODEL + hi * 8;

    v8f acc[2][4] = {{v8f{}, v8f{}, v8f{}, v8f{}}, {v8f{}, v8f{}, v8f{}, v8f{}}};

    v16h a_cur[2], b_cur[4], a_nxt[2], b_nxt[4];
    a_cur[0] = load_afrag(arow0);
    a_cur[1] = load_afrag(arow1);
#pragma unroll
    for (int nt = 0; nt < 4; ++nt) b_cur[nt] = load_bfrag(brow[nt]);

#pragma unroll 2
    for (int k = 0; k < D_MODEL; k += 32) {
        const int kn = (k + 32) & (D_MODEL - 1);
        a_nxt[0] = load_afrag(arow0 + kn);
        a_nxt[1] = load_afrag(arow1 + kn);
#pragma unroll
        for (int nt = 0; nt < 4; ++nt) b_nxt[nt] = load_bfrag(brow[nt] + kn);
        __builtin_prefetch(arow0 + ((k + 128) & (D_MODEL - 1)), 0, 0);

#pragma unroll
        for (int nt = 0; nt < 4; ++nt) acc[0][nt] = wmma16(a_cur[0], b_cur[nt], acc[0][nt]);
#pragma unroll
        for (int nt = 0; nt < 4; ++nt) acc[1][nt] = wmma16(a_cur[1], b_cur[nt], acc[1][nt]);

        a_cur[0] = a_nxt[0]; a_cur[1] = a_nxt[1];
#pragma unroll
        for (int nt = 0; nt < 4; ++nt) b_cur[nt] = b_nxt[nt];
    }

#pragma unroll
    for (int nt = 0; nt < 4; ++nt) {
        const int n    = Nbase + nt * 16 + col;
        const float bv = bias[n];
#pragma unroll
        for (int mt = 0; mt < 2; ++mt) {
#pragma unroll
            for (int r = 0; r < 8; ++r) {
                const int token = Mbase + mt * 16 + r + hi * 8;
                out[(size_t)token * D_MODEL + n] = acc[mt][nt][r] + bv;
            }
        }
    }
}

// --------------------------------------------------------------------------
// Host-side orchestration
// --------------------------------------------------------------------------
extern "C" void kernel_launch(void* const* d_in, const int* in_sizes, int n_in,
                              void* d_out, int out_size, void* d_ws, size_t ws_size,
                              hipStream_t stream) {
    (void)in_sizes; (void)n_in; (void)out_size; (void)ws_size;

    const float* x      = (const float*)d_in[0];  // [2,2048,1024]
    const float* qkv_w  = (const float*)d_in[1];  // [3072,1024]
    const float* qkv_b  = (const float*)d_in[2];  // [3072]
    const float* proj_w = (const float*)d_in[3];  // [1024,1024]
    const float* proj_b = (const float*)d_in[4];  // [1024]
    float* out = (float*)d_out;                   // [4096,1024]

    const int NT  = 2 * L_SEQ;             // 4096 tokens
    const int N3D = 3 * D_MODEL;           // 3072

    char* ws = (char*)d_ws;
    size_t off = 0;
    auto carve = [&](size_t bytes) -> char* {
        char* p = ws + off;
        off += (bytes + 255) & ~(size_t)255;
        return p;
    };
    _Float16* Xh    = (_Float16*)carve((size_t)NT * D_MODEL * 2);      // 8 MB
    _Float16* Wqkvh = (_Float16*)carve((size_t)N3D * D_MODEL * 2);     // 6 MB
    _Float16* Wprjh = (_Float16*)carve((size_t)D_MODEL * D_MODEL * 2); // 2 MB
    _Float16* Qh    = (_Float16*)carve((size_t)NT * D_MODEL * 2);      // 8 MB
    _Float16* Kh    = (_Float16*)carve((size_t)NT * D_MODEL * 2);      // 8 MB
    _Float16* Vt    = (_Float16*)carve((size_t)NT * D_MODEL * 2);      // 8 MB
    _Float16* Oh    = Xh;  // Xh is dead after the QKV GEMM; reuse for attn output

    // 1) stage f16 copies
    {
        int n;
        n = NT * D_MODEL;
        cvt_f32_f16_kernel<<<dim3((n / 4 + 255) / 256), dim3(256), 0, stream>>>(x, Xh, n);
        n = N3D * D_MODEL;
        cvt_f32_f16_kernel<<<dim3((n / 4 + 255) / 256), dim3(256), 0, stream>>>(qkv_w, Wqkvh, n);
        n = D_MODEL * D_MODEL;
        cvt_f32_f16_kernel<<<dim3((n / 4 + 255) / 256), dim3(256), 0, stream>>>(proj_w, Wprjh, n);
    }

    // 2) QKV projection -> Qh/Kh/Vt (f16, head-separated; V transposed)
    {
        dim3 grid(N3D / 64, NT / 256);
        qkv_gemm_kernel<<<grid, dim3(256), 0, stream>>>(Xh, Wqkvh, qkv_b, Qh, Kh, Vt);
    }

    // 3) flash attention -> Oh [token, D] f16
    {
        dim3 grid(L_SEQ / 64, 2 * N_HEADS);
        attn_kernel<<<grid, dim3(128), 0, stream>>>(Qh, Kh, Vt, Oh);
    }

    // 4) output projection -> d_out (f32)
    {
        dim3 grid(D_MODEL / 64, NT / 256);
        proj_gemm_kernel<<<grid, dim3(256), 0, stream>>>(Oh, Wprjh, proj_b, out);
    }
}
